// QuantLinear_53850299957966
// MI455X (gfx1250) — matrix operand order repaired and checked by reading, not compiled, not run on hardware
//
#include <hip/hip_runtime.h>
#include <hip/hip_bf16.h>

typedef __attribute__((ext_vector_type(16))) _Float16 v16h;
typedef __attribute__((ext_vector_type(8)))  float    v8f;

#define DIN   4096
#define DOUT  12288
#define NTOK  16
#define GROUP 128
#define WAVES_PER_BLOCK 8   // 256 threads, wave32 -> 8 waves

// Each wave computes a 16(M=tokens) x 16(N=out cols) f32 tile, iterating K.
// blockIdx.y = K-slice (split-K). dst has one [16 x 12288] panel per slice.
__global__ __launch_bounds__(256) void qlinear_wmma_gemm(
    const float* __restrict__ x,      // [16, 4096]
    const float* __restrict__ w,      // [12288, 4096]
    const float* __restrict__ scales, // [393216]
    const float* __restrict__ zeros,
    const float* __restrict__ dzero,
    const float* __restrict__ smin,
    const float* __restrict__ smax,
    float* __restrict__ dst,          // [KS][16][12288] (or [16][12288] if KS==1)
    const float* __restrict__ bias,   // non-null only when KS==1
    int klen)                         // K per slice (multiple of 128)
{
    const int lane = threadIdx.x & 31;
    const int wave = threadIdx.x >> 5;
    const int ncol = lane & 15;            // 0..15 within tile
    const bool hiHalf = lane >= 16;

    const int nbase = blockIdx.x * (WAVES_PER_BLOCK * 16) + wave * 16;
    const int o     = nbase + ncol;        // weight row == output column
    const int kbeg  = blockIdx.y * klen;

    const float* __restrict__ xrow = x + (size_t)ncol * DIN;  // token row m=ncol
    const float* __restrict__ wrow = w + (size_t)o    * DIN;

    const int kAoff = hiHalf ? 16 : 0;     // A-fragment: 16 contiguous K per lane half
    const int kBoff = hiHalf ? 16 : 0;     // B-fragment: 16 contiguous K per lane half

    v8f acc = {};

    for (int kg = kbeg; kg < kbeg + klen; kg += GROUP) {
        // ---- per-group quant params (whole 128-chunk shares one group) ----
        const int g = (o << 5) + (kg >> 7);            // o*32 + kg/128
        float s  = scales[g];
        s        = fminf(fmaxf(s, smin[g]), smax[g]);  // clamp_ste(scales)
        float z  = zeros[g];
        float d  = fminf(fmaxf(dzero[g], -s), s);      // clamp_ste(delta_zeros)
        const float rs = 1.0f / s;
        const float c2 = fmaf(z, s, d);                // zeros*scale + sdz
        const float nc2 = -c2;

        // prefetch next weight chunk for this lane's row (global_prefetch_b8)
        if (kg + GROUP < kbeg + klen)
            __builtin_prefetch(wrow + kg + GROUP + kBoff, 0, 0);

        #pragma unroll
        for (int kk = 0; kk < GROUP; kk += 32) {
            const int k0 = kg + kk;

            // ---- A fragment: x[m, k0..k0+31] in f16, contiguous K per lane half (k = 16h + s) ----
            const float* pa = xrow + k0 + kAoff;
            const float4 a0 = *(const float4*)(pa);
            const float4 a1 = *(const float4*)(pa + 4);
            const float4 a2 = *(const float4*)(pa + 8);
            const float4 a3 = *(const float4*)(pa + 12);
            v16h afrag;
            afrag[0]  = (_Float16)a0.x; afrag[1]  = (_Float16)a0.y;
            afrag[2]  = (_Float16)a0.z; afrag[3]  = (_Float16)a0.w;
            afrag[4]  = (_Float16)a1.x; afrag[5]  = (_Float16)a1.y;
            afrag[6]  = (_Float16)a1.z; afrag[7]  = (_Float16)a1.w;
            afrag[8]  = (_Float16)a2.x; afrag[9]  = (_Float16)a2.y;
            afrag[10] = (_Float16)a2.z; afrag[11] = (_Float16)a2.w;
            afrag[12] = (_Float16)a3.x; afrag[13] = (_Float16)a3.y;
            afrag[14] = (_Float16)a3.z; afrag[15] = (_Float16)a3.w;

            // ---- B fragment: dequantized w[o, k0..k0+31], 16 contiguous K/lane ----
            const float* pb = wrow + k0 + kBoff;
            v16h bfrag;
            #pragma unroll
            for (int i = 0; i < 4; ++i) {
                const float4 wv = *(const float4*)(pb + i * 4);
                float e0 = fmaf(wv.x, rs, z);
                float e1 = fmaf(wv.y, rs, z);
                float e2 = fmaf(wv.z, rs, z);
                float e3 = fmaf(wv.w, rs, z);
                e0 = fminf(fmaxf(__builtin_rintf(e0), 0.0f), 15.0f);
                e1 = fminf(fmaxf(__builtin_rintf(e1), 0.0f), 15.0f);
                e2 = fminf(fmaxf(__builtin_rintf(e2), 0.0f), 15.0f);
                e3 = fminf(fmaxf(__builtin_rintf(e3), 0.0f), 15.0f);
                bfrag[i * 4 + 0] = (_Float16)fmaf(e0, s, nc2);
                bfrag[i * 4 + 1] = (_Float16)fmaf(e1, s, nc2);
                bfrag[i * 4 + 2] = (_Float16)fmaf(e2, s, nc2);
                bfrag[i * 4 + 3] = (_Float16)fmaf(e3, s, nc2);
            }

            // D = A x B + C  (v_wmma_f32_16x16x32_f16)
            acc = __builtin_amdgcn_wmma_f32_16x16x32_f16(
                false, afrag, false, bfrag, (short)0, acc, false, false);
        }
    }

    // ---- epilogue: D layout -> (m, o); m = vgpr + (hiHalf ? 8 : 0) ----
    float* __restrict__ panel = dst + (size_t)blockIdx.y * (NTOK * DOUT);
    const float bv = bias ? bias[o] : 0.0f;
    const int mbase = hiHalf ? 8 : 0;
    #pragma unroll
    for (int j = 0; j < 8; ++j)
        panel[(size_t)(mbase + j) * DOUT + o] = acc[j] + bv;
}

// Deterministic split-K reduction: out = bias + sum of KS partial panels.
__global__ __launch_bounds__(256) void qlinear_reduce(
    const float* __restrict__ partial, const float* __restrict__ bias,
    float* __restrict__ out, int ks)
{
    const int i = blockIdx.x * blockDim.x + threadIdx.x;
    if (i >= NTOK * DOUT) return;
    const int o = i % DOUT;
    float v = bias[o];
    for (int s = 0; s < ks; ++s)
        v += partial[(size_t)s * (NTOK * DOUT) + i];
    out[i] = v;
}

extern "C" void kernel_launch(void* const* d_in, const int* in_sizes, int n_in,
                              void* d_out, int out_size, void* d_ws, size_t ws_size,
                              hipStream_t stream) {
    const float* x      = (const float*)d_in[0];
    const float* w      = (const float*)d_in[1];
    const float* bias   = (const float*)d_in[2];
    const float* scales = (const float*)d_in[3];
    const float* zeros  = (const float*)d_in[4];
    const float* dzero  = (const float*)d_in[5];
    const float* smin   = (const float*)d_in[6];
    const float* smax   = (const float*)d_in[7];
    float* out = (float*)d_out;

    const int ntile_blocks = DOUT / (WAVES_PER_BLOCK * 16);  // 96
    const int KS = 4;
    const size_t partial_bytes = (size_t)KS * NTOK * DOUT * sizeof(float);

    if (ws_size >= partial_bytes) {
        // Split-K=4: 384 blocks x 8 waves = 3072 waves to saturate 23.3 TB/s.
        float* partial = (float*)d_ws;
        dim3 grid(ntile_blocks, KS);
        qlinear_wmma_gemm<<<grid, 256, 0, stream>>>(
            x, w, scales, zeros, dzero, smin, smax, partial, nullptr, DIN / KS);
        const int n = NTOK * DOUT;
        qlinear_reduce<<<(n + 255) / 256, 256, 0, stream>>>(partial, bias, out, KS);
    } else {
        // Fallback: single pass, bias fused in epilogue.
        dim3 grid(ntile_blocks, 1);
        qlinear_wmma_gemm<<<grid, 256, 0, stream>>>(
            x, w, scales, zeros, dzero, smin, smax, out, bias, DIN);
    }
}
